// Pointnet2SSG_Feature_44040594653942
// MI455X (gfx1250) — matrix operand order repaired and checked by reading, not compiled, not run on hardware
//
#include <hip/hip_runtime.h>
#include <stdint.h>
#include <stddef.h>

typedef __attribute__((ext_vector_type(16))) _Float16 v16h;
typedef __attribute__((ext_vector_type(8)))  float    v8f;
typedef __attribute__((ext_vector_type(4)))  unsigned int u32x4;

#define NBATCH 8

// ---------------- elementwise copy (l_xyz[0]) ----------------
__global__ void pn2_copy_f32(const float* __restrict__ src, float* __restrict__ dst, int n) {
  int i = blockIdx.x * blockDim.x + threadIdx.x;
  if (i < n) dst[i] = src[i];
}

// ---------------- FPS: one block per batch, dists in LDS ----------------
__global__ void pn2_fps(const float* __restrict__ xyz, int N, int npoint,
                        float* __restrict__ newxyz) {
  __shared__ float dists[4096];
  __shared__ float rv[256];
  __shared__ int   ri[256];
  __shared__ float lastp[3];
  __shared__ int   lastIdx;
  const int b = blockIdx.x;
  const int t = threadIdx.x;
  const float* px = xyz + (size_t)b * N * 3;
  float* outp = newxyz + (size_t)b * npoint * 3;
  for (int i = t; i < N; i += 256) dists[i] = 1e10f;
  if (t == 0) {
    lastIdx = 0;
    outp[0] = px[0]; outp[1] = px[1]; outp[2] = px[2];
  }
  __syncthreads();
  for (int it = 1; it < npoint; ++it) {
    if (t == 0) {
      lastp[0] = px[lastIdx*3+0];
      lastp[1] = px[lastIdx*3+1];
      lastp[2] = px[lastIdx*3+2];
    }
    __syncthreads();
    float best = -1.0f; int bi = 0;
    for (int i = t; i < N; i += 256) {
      float dx = px[i*3+0]-lastp[0];
      float dy = px[i*3+1]-lastp[1];
      float dz = px[i*3+2]-lastp[2];
      float d = dx*dx + dy*dy + dz*dz;
      float dm = fminf(dists[i], d);
      dists[i] = dm;
      if (dm > best) { best = dm; bi = i; }
    }
    rv[t] = best; ri[t] = bi;
    __syncthreads();
    for (int ofs = 128; ofs > 0; ofs >>= 1) {
      if (t < ofs) {
        float ov = rv[t+ofs]; int oi = ri[t+ofs];
        if (ov > rv[t] || (ov == rv[t] && oi < ri[t])) { rv[t] = ov; ri[t] = oi; }
      }
      __syncthreads();
    }
    if (t == 0) {
      int nxt = ri[0];
      lastIdx = nxt;
      outp[it*3+0] = px[nxt*3+0];
      outp[it*3+1] = px[nxt*3+1];
      outp[it*3+2] = px[nxt*3+2];
    }
    __syncthreads();
  }
}

// ---------------- ball query: one thread per centroid ----------------
__global__ void pn2_ballquery(const float* __restrict__ newxyz, const float* __restrict__ xyz,
                              int N, int S, float r2, int ns, int* __restrict__ grpIdx) {
  int id = blockIdx.x * blockDim.x + threadIdx.x;
  if (id >= NBATCH * S) return;
  int b = id / S;
  const float* q  = newxyz + (size_t)id * 3;
  const float* px = xyz + (size_t)b * N * 3;
  int* out = grpIdx + (size_t)id * ns;
  float qx = q[0], qy = q[1], qz = q[2];
  int cnt = 0, first = 0;
  for (int j = 0; j < N && cnt < ns; ++j) {
    float dx = px[j*3+0]-qx, dy = px[j*3+1]-qy, dz = px[j*3+2]-qz;
    float d = dx*dx + dy*dy + dz*dz;
    if (d < r2) { if (cnt == 0) first = j; out[cnt++] = j; }
  }
  for (; cnt < ns; ++cnt) out[cnt] = first;
}

// ---------------- grouping -> f16 A-matrix rows [M, Kpad] ----------------
__global__ void pn2_group(const float* __restrict__ xyz, const float* __restrict__ newxyz,
                          const float* __restrict__ feats, const int* __restrict__ grpIdx,
                          _Float16* __restrict__ X, int Mtot, int S, int ns, int N,
                          int Cf, int Kpad) {
  int rid = blockIdx.x * blockDim.x + threadIdx.x;
  if (rid >= Mtot) return;
  int s = (rid / ns) % S;
  int b = rid / (ns * S);
  int gi = grpIdx[rid];
  _Float16* row = X + (size_t)rid * Kpad;
  const float* p = xyz + ((size_t)b * N + gi) * 3;
  const float* q = newxyz + ((size_t)b * S + s) * 3;
  row[0] = (_Float16)(p[0]-q[0]);
  row[1] = (_Float16)(p[1]-q[1]);
  row[2] = (_Float16)(p[2]-q[2]);
  for (int c = 0; c < Cf; ++c)
    row[3+c] = (_Float16)feats[((size_t)b * N + gi) * Cf + c];
  for (int k = 3 + Cf; k < Kpad; ++k) row[k] = (_Float16)0.0f;
}

// ---------------- W [K,C] f32 -> Wt [C,Kpad] f16 (zero-padded) ----------------
__global__ void pn2_wprep(const float* __restrict__ W, _Float16* __restrict__ Wt,
                          int K, int C, int Kpad) {
  int i = blockIdx.x * blockDim.x + threadIdx.x;
  if (i >= C * Kpad) return;
  int c = i / Kpad, k = i % Kpad;
  Wt[i] = (k < K) ? (_Float16)W[(size_t)k * C + c] : (_Float16)0.0f;
}

// ---------------- zero small buffer ----------------
__global__ void pn2_zero(float* __restrict__ p, int n) {
  int i = blockIdx.x * blockDim.x + threadIdx.x;
  if (i < n) p[i] = 0.0f;
}

// ---------------- WMMA GEMM: Y[M,C] = X[M,Kpad] * Wt[C,Kpad]^T ----------------
// One wave32 per 32x32 output tile (2x2 register blocking, 4 independent
// accumulators -> 4 v_wmma per K-step, 2x operand reuse vs a 16x16 tile).
// Fragment layouts per CDNA5 ISA 7.12.2.
__global__ __launch_bounds__(32)
void pn2_gemm_wmma(const _Float16* __restrict__ X, const _Float16* __restrict__ Wt,
                   float* __restrict__ Y, int C, int Kpad) {
  const int tm = blockIdx.x * 32;
  const int tn = blockIdx.y * 32;
  const int lane = threadIdx.x;
  const int half = lane >> 4;      // lanes 16-31: second K half / M+8 rows
  const int l15  = lane & 15;
  v8f acc00 = {}, acc01 = {}, acc10 = {}, acc11 = {};
  // A: row = tm+l15 (+16); lane<16 gets K {0..7,16..23}, lane>=16 gets {8..15,24..31}
  const _Float16* arow0 = X  + (size_t)(tm + l15) * Kpad + half * 8;
  const _Float16* arow1 = arow0 + (size_t)16 * Kpad;
  // B: col = tn+l15 (+16); lane<16 gets K {0..7,16..23}, lane>=16 gets {8..15,24..31}
  const _Float16* brow0 = Wt + (size_t)(tn + l15) * Kpad + half * 8;
  const _Float16* brow1 = brow0 + (size_t)16 * Kpad;
  for (int k = 0; k < Kpad; k += 32) {
    // pull next K-tile of the streaming A operand toward the WGP
    __builtin_prefetch(arow0 + k + 32, 0, 1);
    __builtin_prefetch(arow1 + k + 32, 0, 1);
    union { v16h v; u32x4 q[2]; } a0, a1, b0, b1;
    a0.q[0] = *(const u32x4*)(arow0 + k);        // VGPR0-3
    a0.q[1] = *(const u32x4*)(arow0 + k + 16);   // VGPR4-7
    a1.q[0] = *(const u32x4*)(arow1 + k);
    a1.q[1] = *(const u32x4*)(arow1 + k + 16);
    b0.q[0] = *(const u32x4*)(brow0 + k);        // VGPR0-3
    b0.q[1] = *(const u32x4*)(brow0 + k + 16);   // VGPR4-7
    b1.q[0] = *(const u32x4*)(brow1 + k);
    b1.q[1] = *(const u32x4*)(brow1 + k + 16);
    acc00 = __builtin_amdgcn_wmma_f32_16x16x32_f16(false, a0.v, false, b0.v,
                                                   (short)0, acc00, false, false);
    acc01 = __builtin_amdgcn_wmma_f32_16x16x32_f16(false, a0.v, false, b1.v,
                                                   (short)0, acc01, false, false);
    acc10 = __builtin_amdgcn_wmma_f32_16x16x32_f16(false, a1.v, false, b0.v,
                                                   (short)0, acc10, false, false);
    acc11 = __builtin_amdgcn_wmma_f32_16x16x32_f16(false, a1.v, false, b1.v,
                                                   (short)0, acc11, false, false);
  }
  // C/D layout: VGPR r -> (M = r + 8*half, N = l15)
  const int r0 = tm + half * 8;
  #pragma unroll
  for (int r = 0; r < 8; ++r) {
    Y[(size_t)(r0 + r)      * C + tn + l15]      = acc00[r];
    Y[(size_t)(r0 + r)      * C + tn + 16 + l15] = acc01[r];
    Y[(size_t)(r0 + 16 + r) * C + tn + l15]      = acc10[r];
    Y[(size_t)(r0 + 16 + r) * C + tn + 16 + l15] = acc11[r];
  }
}

// ---------------- per-channel sum / sumsq ----------------
__global__ void pn2_stats(const float* __restrict__ Y, float* __restrict__ sums,
                          long long total, int C) {
  __shared__ float ssum[512];
  __shared__ float ssq[512];
  for (int c = threadIdx.x; c < C; c += blockDim.x) { ssum[c] = 0.0f; ssq[c] = 0.0f; }
  __syncthreads();
  long long stride = (long long)gridDim.x * blockDim.x;
  for (long long i = (long long)blockIdx.x * blockDim.x + threadIdx.x; i < total; i += stride) {
    float v = Y[i];
    int c = (int)(i % C);
    atomicAdd(&ssum[c], v);
    atomicAdd(&ssq[c], v * v);
  }
  __syncthreads();
  for (int c = threadIdx.x; c < C; c += blockDim.x) {
    atomicAdd(&sums[c], ssum[c]);
    atomicAdd(&sums[C + c], ssq[c]);
  }
}

// ---------------- BN scale/shift ----------------
__global__ void pn2_finalize(const float* __restrict__ sums, const float* __restrict__ g,
                             const float* __restrict__ bb, float* __restrict__ ab,
                             float invM, int C) {
  int c = blockIdx.x * blockDim.x + threadIdx.x;
  if (c >= C) return;
  float m = sums[c] * invM;
  float v = sums[C + c] * invM - m * m;
  float s = g[c] * rsqrtf(v + 1e-5f);
  ab[c]     = s;
  ab[C + c] = bb[c] - m * s;
}

// ---------------- normalize + ReLU -> next f16 input ----------------
__global__ void pn2_norm_relu(const float* __restrict__ Y, const float* __restrict__ ab,
                              _Float16* __restrict__ Xn, long long total, int C) {
  long long i = (long long)blockIdx.x * blockDim.x + threadIdx.x;
  if (i >= total) return;
  int c = (int)(i % C);
  float v = ab[c] * Y[i] + ab[C + c];
  Xn[i] = (_Float16)fmaxf(v, 0.0f);
}

// ---------------- normalize + ReLU + maxpool over nsample -> l_f ----------------
__global__ void pn2_norm_maxpool(const float* __restrict__ Y, const float* __restrict__ ab,
                                 float* __restrict__ out, int S, int ns, int C) {
  int id = blockIdx.x * blockDim.x + threadIdx.x;
  if (id >= NBATCH * S * C) return;
  int c = id % C;
  int bs = id / C;                 // b*S + s
  float sc = ab[c], sh = ab[C + c];
  float best = -1e30f;
  const float* yr = Y + (size_t)bs * ns * C + c;
  for (int n = 0; n < ns; ++n) {
    float v = fmaxf(sc * yr[(size_t)n * C] + sh, 0.0f);
    best = fmaxf(best, v);
  }
  out[id] = best;
}

extern "C" void kernel_launch(void* const* d_in, const int* in_sizes, int n_in,
                              void* d_out, int out_size, void* d_ws, size_t ws_size,
                              hipStream_t stream) {
  (void)in_sizes; (void)n_in; (void)out_size; (void)ws_size;
  const float* in_shape = (const float*)d_in[0];   // (8,2,2048,3) -> (8,4096,3)
  float* out = (float*)d_out;

  // ---- module configs ----
  static const int   Ns [4] = {4096, 2048, 512, 128};   // input point count per module
  static const int   NP [4] = {2048, 512, 128, 16};     // npoint
  static const int   NSm[4] = {32, 32, 16, 16};         // nsample
  static const float R2 [4] = {0.0004f, 0.0016f, 0.0036f, 0.0064f};
  static const int   CIN[4] = {3, 67, 131, 259};
  static const int   KP0[4] = {32, 96, 160, 288};       // cin padded to mult of 32
  static const int   DIMS[4][3] = {{32,32,64},{64,64,128},{128,128,256},{256,256,512}};
  static const int   CF [4] = {0, 64, 128, 256};        // feat channels from prev module

  // ---- d_out offsets (floats) ----
  static const size_t xyzOff[5] = {0, 98304, 147456, 159744, 162816};
  static const size_t fOff [4] = {163200, 1211776, 1736064, 1998208};

  // ---- workspace layout (bytes) ----
  char* ws = (char*)d_ws;
  int*      grpIdx = (int*)      (ws + 0);            // 8*2048*32*4 = 2,097,152
  float*    sums   = (float*)    (ws + 2097152);      // 2*512*4
  float*    ab     = (float*)    (ws + 2101248);      // 2*512*4
  _Float16* Wt     = (_Float16*) (ws + 2105344);      // 512*256*2 max -> 262,144 B reserved
  _Float16* Xa     = (_Float16*) (ws + 2367488);      // 33,554,432 B
  _Float16* Xb     = (_Float16*) (ws + 35921920);     // 33,554,432 B
  float*    Yb     = (float*)    (ws + 69476352);     // 134,217,728 B

  // l_xyz[0] = input xyz
  pn2_copy_f32<<<(98304 + 255) / 256, 256, 0, stream>>>(in_shape, out + xyzOff[0], 98304);

  for (int k = 0; k < 4; ++k) {
    const int N = Ns[k], S = NP[k], ns = NSm[k];
    const int M = NBATCH * S * ns;
    const float* xyz_in  = out + xyzOff[k];
    float*       xyz_new = out + xyzOff[k + 1];
    const float* feats = (k == 0) ? nullptr : (out + fOff[k - 1]);

    // FPS -> new_xyz (also l_xyz[k+1])
    pn2_fps<<<NBATCH, 256, 0, stream>>>(xyz_in, N, S, xyz_new);

    // ball query
    pn2_ballquery<<<(NBATCH * S + 255) / 256, 256, 0, stream>>>(
        xyz_new, xyz_in, N, S, R2[k], ns, grpIdx);

    // grouping -> Xa [M, Kpad0]
    pn2_group<<<(M + 255) / 256, 256, 0, stream>>>(
        xyz_in, xyz_new, feats, grpIdx, Xa, M, S, ns, N, CF[k], KP0[k]);

    const _Float16* Xcur = Xa;
    _Float16*       Xnxt = Xb;
    int Kp = KP0[k];
    for (int l = 0; l < 3; ++l) {
      const int K = (l == 0) ? CIN[k] : DIMS[k][l - 1];
      const int C = DIMS[k][l];
      const int pbase = 2 + (k * 3 + l) * 3;
      const float* W  = (const float*)d_in[pbase + 0];
      const float* g  = (const float*)d_in[pbase + 1];
      const float* bb = (const float*)d_in[pbase + 2];

      pn2_wprep<<<(C * Kp + 255) / 256, 256, 0, stream>>>(W, Wt, K, C, Kp);
      pn2_zero<<<(2 * C + 255) / 256, 256, 0, stream>>>(sums, 2 * C);

      dim3 ggrid(M / 32, C / 32, 1);     // 32x32 tile per wave32
      pn2_gemm_wmma<<<ggrid, 32, 0, stream>>>(Xcur, Wt, Yb, C, Kp);

      long long total = (long long)M * C;
      pn2_stats<<<512, 256, 0, stream>>>(Yb, sums, total, C);
      pn2_finalize<<<(C + 255) / 256, 256, 0, stream>>>(sums, g, bb, ab, 1.0f / (float)M, C);

      if (l < 2) {
        pn2_norm_relu<<<(unsigned)((total + 255) / 256), 256, 0, stream>>>(Yb, ab, Xnxt, total, C);
        const _Float16* t = Xcur; Xcur = Xnxt; Xnxt = (_Float16*)t;
        Kp = C;
      } else {
        pn2_norm_maxpool<<<(NBATCH * S * C + 255) / 256, 256, 0, stream>>>(
            Yb, ab, out + fOff[k], S, ns, C);
      }
    }
  }
}
